// VectorQuantizer_65085934403890
// MI455X (gfx1250) — compile-verified, matrix operand order checked by reading, not run on hardware
//
#include <hip/hip_runtime.h>

typedef __attribute__((ext_vector_type(16))) __bf16 v16bf;
typedef __attribute__((ext_vector_type(8)))  float  v8f;

// Problem constants
#define Bn   64
#define Cn   64
#define Hn   32
#define Wn   32
#define Kn   1024
#define Dn   64
#define HWn  (Hn*Wn)        // 1024
#define CHWn (Cn*HWn)       // 65536 floats per image
#define NPIX (Bn*HWn)       // 65536 pixel vectors

// Packed codebook sizes (bf16 elements): Kn*Dn = 65536 per (hi|lo)
#define PACKED_ELEMS (Kn*Dn)          // 65536

// Workspace layout (bytes): [0,128K) bhi | [128K,256K) blo | [256K,256K+4K) e_sq
#define WS_BHI   0
#define WS_BLO   131072
#define WS_ESQ   262144
#define STAGE_BYTES 266240            // bhi + blo + esq (copied to LDS)

// LDS layout: same as workspace, plus per-wave argmin scratch
#define LDS_WIDX 266240
#define LDS_TOTAL (266240 + 16*16*4)  // 267264 bytes

// K-element permutation for the 16-bit A/B WMMA fragment layout (ISA 7.12.2):
// lanes 0-15 hold K = {0..7, 16..23}; lanes 16-31 hold K = {8..15, 24..31}
__device__ __forceinline__ int kperm(int lane, int j) {
    int base = (lane >= 16) ? 8 : 0;
    return base + (j < 8 ? j : j + 8);
}

// ---------------------------------------------------------------------------
// Prep 1: pre-swizzle codebook into WMMA B-fragment order, split bf16 hi/lo.
// Element tid = ((t*2+c)*32 + L)*16 + j maps to e[col, k],
// col = t*16 + (L&15), k = c*32 + kperm(L, j).
// ---------------------------------------------------------------------------
__global__ void vq_prep_pack(const float* __restrict__ emb, char* __restrict__ ws) {
    int tid = blockIdx.x * blockDim.x + threadIdx.x;
    if (tid >= PACKED_ELEMS) return;
    int j = tid & 15;
    int L = (tid >> 4) & 31;
    int c = (tid >> 9) & 1;
    int t = tid >> 10;
    int col = t * 16 + (L & 15);
    int k   = c * 32 + kperm(L, j);
    float v = emb[col * Dn + k];
    __bf16 h = (__bf16)v;
    __bf16 l = (__bf16)(v - (float)h);
    ((__bf16*)(ws + WS_BHI))[tid] = h;
    ((__bf16*)(ws + WS_BLO))[tid] = l;
}

// ---------------------------------------------------------------------------
// Prep 2: e_sq[k] = sum_d e[k,d]^2 ; thread 0 zeroes the loss accumulator.
// ---------------------------------------------------------------------------
__global__ void vq_prep_esq(const float* __restrict__ emb, char* __restrict__ ws,
                            float* __restrict__ out_loss) {
    int k = blockIdx.x * blockDim.x + threadIdx.x;
    if (k == 0) out_loss[0] = 0.0f;
    if (k >= Kn) return;
    float s = 0.0f;
    const float* row = emb + k * Dn;
    #pragma unroll
    for (int d = 0; d < Dn; ++d) s += row[d] * row[d];
    ((float*)(ws + WS_ESQ))[k] = s;
}

// ---------------------------------------------------------------------------
// Main kernel: 512 threads (16 waves), 256 blocks. Each wave owns 16 pixel
// rows, sweeps all 1024 codes with split-bf16 WMMA (3 independent accumulator
// chains, double-buffered LDS B-fragments), argmins, gathers, writes the
// quantized NCHW output and accumulates the fused loss.
// ---------------------------------------------------------------------------
__global__ __launch_bounds__(512, 1)
void vq_main(const float* __restrict__ x, const float* __restrict__ emb,
             const char* __restrict__ ws, float* __restrict__ out) {
    extern __shared__ char smem[];
    const int tid  = threadIdx.x;
    const int lane = tid & 31;
    const int wv   = tid >> 5;

    // ---- Stage codebook (hi/lo bf16, 256 KB) + e_sq (4 KB) into LDS via the
    //      async global->LDS path (no VGPR round-trip, tracked by ASYNCcnt).
    //      No static __shared__ in this kernel => dynamic LDS starts at 0.
    {
        const unsigned long long wsbase = (unsigned long long)ws;
        for (int i = tid; i < (STAGE_BYTES / 16); i += 512) {
            unsigned ldsa = (unsigned)i * 16u;   // LDS byte offset
            unsigned voff = (unsigned)i * 16u;   // global byte offset
            asm volatile("global_load_async_to_lds_b128 %0, %1, %2"
                         :: "v"(ldsa), "v"(voff), "s"(wsbase) : "memory");
        }
        asm volatile("s_wait_asynccnt 0" ::: "memory");
    }
    __syncthreads();

    // ---- Load this wave's 16 x-rows as A fragments (hi/lo), kept in VGPRs ----
    const int n0  = (blockIdx.x * 16 + wv) * 16;
    const int row = n0 + (lane & 15);
    const int bi  = row >> 10;        // batch
    const int hw  = row & 1023;       // h*W + w
    const float* xrow = x + (size_t)bi * CHWn + hw;   // element d at +d*HWn

    v16bf ah[2], al[2];
    #pragma unroll
    for (int c = 0; c < 2; ++c) {
        #pragma unroll
        for (int j = 0; j < 16; ++j) {
            int d  = c * 32 + kperm(lane, j);
            float f = xrow[(size_t)d * HWn];
            __bf16 h = (__bf16)f;
            ah[c][j] = h;
            al[c][j] = (__bf16)(f - (float)h);
        }
    }

    const char*  bhi  = smem + WS_BHI;
    const char*  blo  = smem + WS_BLO;
    const float* esq  = (const float*)(smem + WS_ESQ);
    int*         widx = (int*)(smem + LDS_WIDX);

    // ---- Sweep all 64 column tiles: score = e_sq - 2*x.e, track argmin ----
    float bestv[8];
    int   besti[8];
    #pragma unroll
    for (int r = 0; r < 8; ++r) { bestv[r] = 3.4e38f; besti[r] = 0; }
    const int colbase = lane & 15;

    auto loadTile = [&](int t, v16bf (&bh)[2], v16bf (&bl)[2]) {
        #pragma unroll
        for (int c = 0; c < 2; ++c) {
            size_t off = ((size_t)(t * 2 + c) * 32 + lane) * 32;
            bh[c] = *(const v16bf*)(bhi + off);
            bl[c] = *(const v16bf*)(blo + off);
        }
    };
    auto computeTile = [&](int t, const v16bf (&bh)[2], const v16bf (&bl)[2]) {
        // split-bf16 fp32 emulation, 3 independent WMMA chains for pipelining
        v8f hh = {}, lh = {}, hl = {};
        #pragma unroll
        for (int c = 0; c < 2; ++c) {
            hh = __builtin_amdgcn_wmma_f32_16x16x32_bf16(false, ah[c], false, bh[c],
                                                         (short)0, hh, false, false);
            lh = __builtin_amdgcn_wmma_f32_16x16x32_bf16(false, al[c], false, bh[c],
                                                         (short)0, lh, false, false);
            hl = __builtin_amdgcn_wmma_f32_16x16x32_bf16(false, ah[c], false, bl[c],
                                                         (short)0, hl, false, false);
        }
        int   col = t * 16 + colbase;
        float es  = esq[col];
        #pragma unroll
        for (int r = 0; r < 8; ++r) {
            float s = es - 2.0f * ((hh[r] + lh[r]) + hl[r]);
            if (s < bestv[r]) { bestv[r] = s; besti[r] = col; }
        }
    };

    // Double-buffered register pipeline over the 64 column tiles
    v16bf bhA[2], blA[2], bhB[2], blB[2];
    loadTile(0, bhA, blA);
    for (int tt = 0; tt < 32; ++tt) {
        int t0 = tt * 2;
        loadTile(t0 + 1, bhB, blB);
        computeTile(t0, bhA, blA);
        if (tt < 31) loadTile(t0 + 2, bhA, blA);
        computeTile(t0 + 1, bhB, blB);
    }

    // ---- Cross-lane argmin over the 16 N-lanes (rows 0-7 in lanes 0-15,
    //      rows 8-15 in lanes 16-31); ties -> lowest index like jnp.argmin ----
    #pragma unroll
    for (int m = 8; m >= 1; m >>= 1) {
        #pragma unroll
        for (int r = 0; r < 8; ++r) {
            float ov = __shfl_xor(bestv[r], m, 32);
            int   oi = __shfl_xor(besti[r], m, 32);
            if (ov < bestv[r] || (ov == bestv[r] && oi < besti[r])) {
                bestv[r] = ov; besti[r] = oi;
            }
        }
    }
    float* out_idx = out + 1 + (size_t)Bn * CHWn;   // after loss + quantized
    if ((lane & 15) == 0) {
        int half = lane >> 4;   // 0: rows 0-7, 1: rows 8-15
        #pragma unroll
        for (int r = 0; r < 8; ++r) {
            int rr = r + half * 8;
            widx[wv * 16 + rr] = besti[r];
            out_idx[n0 + rr]   = (float)besti[r];
        }
    }
    __syncthreads();

    // ---- Gather e[idx], write quantized (NCHW) + fused loss ----
    const int rr  = lane & 15;                 // row within wave tile
    const int par = lane >> 4;                 // channel parity
    const int n   = n0 + rr;
    const int kk  = widx[wv * 16 + rr];
    const int b2  = n >> 10;
    const int hw2 = n & 1023;
    const float* xp = x + (size_t)b2 * CHWn + hw2;
    float*       qp = out + 1 + (size_t)b2 * CHWn + hw2;
    const float* ep = emb + (size_t)kk * Dn;

    float lsum = 0.0f;
    #pragma unroll
    for (int dd = 0; dd < 32; ++dd) {
        int d = dd * 2 + par;
        float q  = ep[d];
        float xv = xp[(size_t)d * HWn];
        qp[(size_t)d * HWn] = q;
        float df = q - xv;
        lsum += df * df;
    }
    #pragma unroll
    for (int m = 16; m >= 1; m >>= 1) lsum += __shfl_xor(lsum, m, 32);
    // loss = (1 + 0.25) * mean(diff^2) over B*H*W*D = 4194304 elements
    if (lane == 0) atomicAdd(out, lsum * (1.25f / 4194304.0f));
}

// ---------------------------------------------------------------------------
extern "C" void kernel_launch(void* const* d_in, const int* in_sizes, int n_in,
                              void* d_out, int out_size, void* d_ws, size_t ws_size,
                              hipStream_t stream) {
    (void)in_sizes; (void)n_in; (void)out_size; (void)ws_size;
    const float* x   = (const float*)d_in[0];
    const float* emb = (const float*)d_in[1];
    char*  ws  = (char*)d_ws;
    float* out = (float*)d_out;

    hipFuncSetAttribute((const void*)vq_main,
                        hipFuncAttributeMaxDynamicSharedMemorySize, LDS_TOTAL);

    vq_prep_pack<<<PACKED_ELEMS / 256, 256, 0, stream>>>(emb, ws);
    vq_prep_esq<<<Kn / 256, 256, 0, stream>>>(emb, ws, out);
    vq_main<<<NPIX / 256, 512, LDS_TOTAL, stream>>>(x, emb, ws, out);
}
